// HyperAttention_23218593202740
// MI455X (gfx1250) — matrix operand order repaired and checked by reading, not compiled, not run on hardware
//
#include <hip/hip_runtime.h>
#include <hip/hip_bf16.h>

// ---------------------------------------------------------------------------
// HyperAttention fused pipeline for MI455X (gfx1250, wave32, WMMA + async-LDS)
// A=1024, C=CP=128, H=8, kph=16
// ---------------------------------------------------------------------------

#define A_DIM  1024
#define C_DIM  128
#define H_DIM  8
#define EPSF   1e-6f

typedef __attribute__((ext_vector_type(8)))  float   v8f;
typedef __attribute__((ext_vector_type(16))) __bf16  v16bf;
typedef __attribute__((ext_vector_type(8)))  __bf16  v8bf;

__device__ inline v16bf make_v16(v8bf lo, v8bf hi) {
    return __builtin_shufflevector(lo, hi, 0,1,2,3,4,5,6,7,8,9,10,11,12,13,14,15);
}

__device__ inline v8f wmma_bf16(v16bf a, v16bf b, v8f c) {
    return __builtin_amdgcn_wmma_f32_16x16x32_bf16(
        /*neg_a=*/false, a, /*neg_b=*/false, b,
        /*c_mod=*/(short)0, c, /*reuse_a=*/false, /*reuse_b=*/false);
}

// A-matrix 16x32 bf16 fragment from row-major [16 x ld] buffer (LDS or global).
// ISA layout: lane m=lane&15, hi=lane>>4; e<8 -> k=kbase+e+8*hi,
// e>=8 -> k=kbase+16+(e-8)+8*hi  => two contiguous 8-elem (16B) loads.
__device__ inline v16bf load_a_frag(const __bf16* p, int ld, int kbase, int lane) {
    int m = lane & 15, hi = lane >> 4;
    const __bf16* r = p + m * ld + kbase + 8 * hi;
    v8bf lo  = *(const v8bf*)(r);
    v8bf hi8 = *(const v8bf*)(r + 16);
    return make_v16(lo, hi8);
}

__device__ inline float red_sum16(float v) {
    #pragma unroll
    for (int off = 8; off; off >>= 1) v += __shfl_xor(v, off, 16);
    return v;
}
__device__ inline float red_max16(float v) {
    #pragma unroll
    for (int off = 8; off; off >>= 1) v = fmaxf(v, __shfl_xor(v, off, 16));
    return v;
}

// Per-lane async copy of 32 contiguous bytes (global -> LDS), ASYNCcnt-tracked.
// INST_OFFSET is added to both the LDS and global address (ISA 15.18.3).
__device__ inline void async_copy_32B(unsigned int lds_off, const float* gsrc) {
    unsigned long long ga = (unsigned long long)(uintptr_t)gsrc;
    asm volatile("global_load_async_to_lds_b128 %0, %1, off"
                 :: "v"(lds_off), "v"(ga) : "memory");
    asm volatile("global_load_async_to_lds_b128 %0, %1, off offset:16"
                 :: "v"(lds_off), "v"(ga) : "memory");
}

// ---------------------------------------------------------------------------
// Utility kernels
// ---------------------------------------------------------------------------
__global__ void zero_kernel(float* p, int n) {
    int i = blockIdx.x * blockDim.x + threadIdx.x;
    for (; i < n; i += gridDim.x * blockDim.x) p[i] = 0.f;
}

// Swizzle a [128 x ncols] fp32 weight into bf16 WMMA B-fragment order:
// out[((ks*ntiles + nt)*32 + lane)*16 + e] = w[k*ld + n],
//   k = ks*32 + 8*(lane>>4) + (e<8 ? e : 16+e-8),  n = nt*16 + (lane&15)   (zero-padded)
__global__ void swizzle_w_kernel(const float* __restrict__ w, __bf16* __restrict__ out,
                                 int ld, int ncols, int ntiles) {
    int flat = blockIdx.x * blockDim.x + threadIdx.x;
    int total = 4 * ntiles * 32 * 16;
    if (flat >= total) return;
    int e    = flat & 15;
    int lane = (flat >> 4) & 31;
    int rest = flat >> 9;
    int nt   = rest % ntiles;
    int ks   = rest / ntiles;
    int k = ks * 32 + 8 * (lane >> 4) + (e < 8 ? e : 16 + (e - 8));
    int n = nt * 16 + (lane & 15);
    out[flat] = (n < ncols) ? (__bf16)w[k * ld + n] : (__bf16)0.f;
}

// LN of atom_act -> x fp32 [1024][128].  64 blocks x 256 threads (16 rows each).
__global__ __launch_bounds__(256)
void ln_x_kernel(const float* __restrict__ act, const float* __restrict__ sc,
                 const float* __restrict__ bi, float* __restrict__ xo) {
    int t   = threadIdx.x;
    int row = blockIdx.x * 16 + (t >> 4);
    int sub = t & 15;
    float vals[8], s = 0.f, s2 = 0.f;
    #pragma unroll
    for (int e = 0; e < 8; ++e) {
        float f = act[row * C_DIM + sub * 8 + e];
        vals[e] = f; s += f; s2 += f * f;
    }
    s = red_sum16(s); s2 = red_sum16(s2);
    float mean = s * (1.f / C_DIM);
    float var  = s2 * (1.f / C_DIM) - mean * mean;
    float rstd = rsqrtf(var + EPSF);
    #pragma unroll
    for (int e = 0; e < 8; ++e) {
        int c = sub * 8 + e;
        xo[row * C_DIM + c] = (vals[e] - mean) * rstd * sc[c] + bi[c];
    }
}

// Row sums (msum1[i]) and column sums (msum0[j]) of pair_mask. 1024 blocks.
__global__ __launch_bounds__(256)
void mask_sum_kernel(const float* __restrict__ pm, float* __restrict__ msum1,
                     float* __restrict__ msum0) {
    int i = blockIdx.x, t = threadIdx.x;
    __shared__ float red[256];
    float s = 0.f;
    for (int j = t; j < A_DIM; j += 256) {
        float m = pm[(size_t)i * A_DIM + j];
        s += m;
        atomicAdd(&msum0[j], m);
    }
    red[t] = s; __syncthreads();
    #pragma unroll
    for (int off = 128; off; off >>= 1) {
        if (t < off) red[t] += red[t + off];
        __syncthreads();
    }
    if (t == 0) msum1[i] = red[0];
}

// ---------------------------------------------------------------------------
// Fused pair pipeline: block = (i, group of 4 j-tiles), 256 threads (8 waves).
// Double-buffered async global->LDS streaming of the raw fp32 tiles, then per
// tile: LN -> bf16 ; WMMA p@w2d -> pair_bias ; WMMA p@w_pm + b_pm -> LN ->
// mask -> accumulate u (registers, one atomic at end) and vmix (atomics).
// grid (1024, 16)
// ---------------------------------------------------------------------------
__global__ __launch_bounds__(256)
void pair_kernel(const float* __restrict__ pair_act,
                 const float* __restrict__ pair_mask,
                 const float* __restrict__ f2n_s, const float* __restrict__ f2n_b,
                 const float* __restrict__ mn_s,  const float* __restrict__ mn_b,
                 const float* __restrict__ b_pm,
                 const __bf16* __restrict__ w2d_f,   // swizzled [4][1][32][16]
                 const __bf16* __restrict__ wpm_f,   // swizzled [4][8][32][16]
                 float* __restrict__ pair_bias,      // [8][1024][1024]
                 float* __restrict__ u_acc,          // [1024][128]
                 float* __restrict__ v_acc)          // [1024][128]
{
    __shared__ float  sRaw[2][16 * 128];  // raw fp32 tiles (async ring buffer)
    __shared__ __bf16 sP[16 * 128];       // LN'd p tile (bf16, A-operand source)
    __shared__ float  sT[16 * 128];       // pair_matrix tile (f32)

    const int i    = blockIdx.x;
    const int jt4  = blockIdx.y;          // group of 4 tiles
    const int t    = threadIdx.x;
    const int lane = t & 31;
    const int wave = t >> 5;
    const int row  = t >> 4;              // 0..15 (j within tile)
    const int sub  = t & 15;

    // hoisted weight B-fragments (reused for all 4 tiles)
    v16bf bw[4];
    #pragma unroll
    for (int ks = 0; ks < 4; ++ks)
        bw[ks] = *(const v16bf*)(wpm_f + ((size_t)((ks * 8 + wave) * 32 + lane)) * 16);
    v16bf b2[4] = {};
    if (wave == 0) {
        #pragma unroll
        for (int ks = 0; ks < 4; ++ks)
            b2[ks] = *(const v16bf*)(w2d_f + ((size_t)(ks * 32 + lane)) * 16);
    }

    const size_t base4 = ((size_t)i * A_DIM + (size_t)jt4 * 64) * C_DIM;
    const unsigned int ldsRaw[2] = {
        (unsigned int)(uintptr_t)(&sRaw[0][0]) + (unsigned int)t * 32u,
        (unsigned int)(uintptr_t)(&sRaw[1][0]) + (unsigned int)t * 32u };

    // prologue: stream tile 0
    async_copy_32B(ldsRaw[0], pair_act + base4 + (size_t)t * 8);

    float u_part = 0.f;   // per-thread (t<128) partial of u[i][t]

    for (int c = 0; c < 4; ++c) {
        // stream tile c+1 while computing tile c
        if (c + 1 < 4)
            async_copy_32B(ldsRaw[(c + 1) & 1],
                           pair_act + base4 + (size_t)(c + 1) * 16 * C_DIM + (size_t)t * 8);
        if (c + 1 < 4) asm volatile("s_wait_asynccnt 0x2" ::: "memory");
        else           asm volatile("s_wait_asynccnt 0x0" ::: "memory");

        const float* raw = sRaw[c & 1];   // this thread reads only its own bytes

        // ---- phase 1: LayerNorm of 16x128 tile -> bf16 in LDS ----
        float vals[8], s = 0.f, s2 = 0.f;
        #pragma unroll
        for (int e = 0; e < 8; ++e) {
            float f = raw[t * 8 + e];
            vals[e] = f; s += f; s2 += f * f;
        }
        s = red_sum16(s); s2 = red_sum16(s2);
        {
            float mean = s * (1.f / C_DIM);
            float var  = s2 * (1.f / C_DIM) - mean * mean;
            float rstd = rsqrtf(var + EPSF);
            #pragma unroll
            for (int e = 0; e < 8; ++e) {
                int cc = sub * 8 + e;
                sP[row * C_DIM + cc] =
                    (__bf16)((vals[e] - mean) * rstd * f2n_s[cc] + f2n_b[cc]);
            }
        }
        __syncthreads();

        const int jb = jt4 * 64 + c * 16;   // global j base of this tile

        // ---- phase 2: pair_bias = p @ w2d  (wave 0; 8 valid columns) ----
        if (wave == 0) {
            v8f acc = {};
            #pragma unroll
            for (int ks = 0; ks < 4; ++ks) {
                v16bf a = load_a_frag(sP, C_DIM, ks * 32, lane);
                acc = wmma_bf16(a, b2[ks], acc);
            }
            int n = lane & 15, hi = lane >> 4;
            if (n < H_DIM) {
                #pragma unroll
                for (int r = 0; r < 8; ++r) {
                    int m = r + 8 * hi;
                    pair_bias[(size_t)n * A_DIM * A_DIM + (size_t)i * A_DIM + jb + m] = acc[r];
                }
            }
        }

        // ---- phase 3: pair_matrix tile = p @ w_pm + b_pm ----
        {
            v8f acc = {};
            const int nb = wave * 16;
            #pragma unroll
            for (int ks = 0; ks < 4; ++ks) {
                v16bf a = load_a_frag(sP, C_DIM, ks * 32, lane);
                acc = wmma_bf16(a, bw[ks], acc);
            }
            int n = lane & 15, hi = lane >> 4;
            #pragma unroll
            for (int r = 0; r < 8; ++r) {
                int m = r + 8 * hi;
                sT[m * C_DIM + nb + n] = acc[r] + b_pm[nb + n];
            }
        }
        __syncthreads();

        // ---- phase 4: LN over d per row, mask, accumulate vmix ----
        {
            float v2[8]; float ss = 0.f, ss2 = 0.f;
            #pragma unroll
            for (int e = 0; e < 8; ++e) {
                float f = sT[row * C_DIM + sub * 8 + e];
                v2[e] = f; ss += f; ss2 += f * f;
            }
            ss = red_sum16(ss); ss2 = red_sum16(ss2);
            float mean = ss * (1.f / C_DIM);
            float var  = ss2 * (1.f / C_DIM) - mean * mean;
            float rstd = rsqrtf(var + EPSF);
            float pm   = pair_mask[(size_t)i * A_DIM + jb + row];
            size_t jg  = (size_t)(jb + row);
            #pragma unroll
            for (int e = 0; e < 8; ++e) {
                int cc = sub * 8 + e;
                float f = ((v2[e] - mean) * rstd * mn_s[cc] + mn_b[cc]) * pm;
                sT[row * C_DIM + cc] = f;
                atomicAdd(&v_acc[jg * C_DIM + cc], f);
            }
        }
        __syncthreads();

        if (t < C_DIM) {   // u partial: sum over the 16 j of this tile
            #pragma unroll
            for (int j = 0; j < 16; ++j) u_part += sT[j * C_DIM + t];
        }
        __syncthreads();   // sT reused next iteration
    }

    if (t < C_DIM) atomicAdd(&u_acc[(size_t)i * C_DIM + t], u_part);
}

// ---------------------------------------------------------------------------
// mix: query_act / key_act (bf16) from x, u, vmix and mask sums
// ---------------------------------------------------------------------------
__global__ __launch_bounds__(256)
void mix_kernel(const float* __restrict__ x, const float* __restrict__ u,
                const float* __restrict__ vm, const float* __restrict__ msum1,
                const float* __restrict__ msum0,
                __bf16* __restrict__ qa_bf, __bf16* __restrict__ ka_bf) {
    int idx = blockIdx.x * 256 + threadIdx.x;       // 0 .. 1024*128-1
    int a = idx >> 7;
    float xx = x[idx];
    float un = u[idx]  / (msum1[a] + EPSF);
    float vn = vm[idx] / (msum0[a] + EPSF);
    qa_bf[idx] = (__bf16)(xx + xx * un);
    ka_bf[idx] = (__bf16)(xx + xx * vn);
}

// ---------------------------------------------------------------------------
// Generic 1024x128 @ 128x128 WMMA GEMM with epilogue.
// grid 64 (16-row tiles), 8 waves = 8 column tiles.
// ---------------------------------------------------------------------------
__global__ __launch_bounds__(256)
void gemm_proj_kernel(const __bf16* __restrict__ Abf,     // [1024][128] row-major
                      const __bf16* __restrict__ Bfrag,   // swizzled [4][8][32][16]
                      const float* __restrict__ bias,     // may be null
                      float scale, int do_sigmoid,
                      __bf16* __restrict__ out_bf,        // may be null
                      __bf16* __restrict__ out_bf_t,      // may be null, [128][1024]
                      float* __restrict__ out_f)          // may be null
{
    const int mt   = blockIdx.x;
    const int wave = threadIdx.x >> 5;
    const int lane = threadIdx.x & 31;
    v8f acc = {};
    #pragma unroll
    for (int ks = 0; ks < 4; ++ks) {
        v16bf a = load_a_frag(Abf + (size_t)mt * 16 * C_DIM, C_DIM, ks * 32, lane);
        v16bf b = *(const v16bf*)(Bfrag + ((size_t)((ks * 8 + wave) * 32 + lane)) * 16);
        acc = wmma_bf16(a, b, acc);
    }
    int n = lane & 15, hi = lane >> 4;
    int col = wave * 16 + n;
    float bb = bias ? bias[col] : 0.f;
    #pragma unroll
    for (int r = 0; r < 8; ++r) {
        int m = mt * 16 + r + 8 * hi;
        float v = acc[r] * scale + bb;
        if (do_sigmoid) v = 1.f / (1.f + __expf(-v));
        if (out_f)    out_f[(size_t)m * C_DIM + col]    = v;
        if (out_bf)   out_bf[(size_t)m * C_DIM + col]   = (__bf16)v;
        if (out_bf_t) out_bf_t[(size_t)col * A_DIM + m] = (__bf16)v;
    }
}

// ---------------------------------------------------------------------------
// Flash-style attention: block per 16-row query tile, wave w = head w.
// Streams 64 key tiles; online softmax; O += P@V via LDS round-trip of P.
// V is consumed transposed ([C][A]) so its B-fragment is one 16B vector load.
// ---------------------------------------------------------------------------
__global__ __launch_bounds__(256)
void attn_kernel(const __bf16* __restrict__ q_bf,   // pre-scaled by kph^-0.5
                 const __bf16* __restrict__ k_bf,
                 const __bf16* __restrict__ v_t,    // [128][1024] transposed
                 const float* __restrict__ pair_bias,  // [8][1024][1024]
                 const float* __restrict__ gate,       // [1024][128] f32
                 __bf16* __restrict__ wa_bf)           // [1024][128]
{
    __shared__ __bf16 sP[H_DIM][16 * 16];
    const int it   = blockIdx.x;            // 0..63
    const int h    = threadIdx.x >> 5;      // head = wave
    const int lane = threadIdx.x & 31;
    const int n  = lane & 15;
    const int hi = lane >> 4;

    // Q fragment: rows m = lane&15, K dim = kph(16) padded to 32 with zeros.
    v8bf z8 = {};
    v16bf aq;
    {
        const __bf16* qp = q_bf + ((size_t)(it * 16 + n)) * C_DIM + h * 16 + 8 * hi;
        v8bf lo = *(const v8bf*)qp;
        aq = make_v16(lo, z8);
    }

    v8f o = {};
    float run_max[8], run_sum[8];
    #pragma unroll
    for (int r = 0; r < 8; ++r) { run_max[r] = -1e30f; run_sum[r] = 0.f; }

    for (int jt = 0; jt < A_DIM / 16; ++jt) {
        // K fragment: B[k][n] = K[jt*16+n][h*16+k], k<16 valid (elements 8..15 zero pad)
        v16bf bk = make_v16(*(const v8bf*)(k_bf + ((size_t)(jt * 16 + n)) * C_DIM + h * 16 + 8 * hi), z8);
        v8f sacc = {};
        sacc = wmma_bf16(aq, bk, sacc);

        #pragma unroll
        for (int r = 0; r < 8; ++r) {
            int m = r + 8 * hi;
            sacc[r] += pair_bias[(size_t)h * A_DIM * A_DIM +
                                 (size_t)(it * 16 + m) * A_DIM + jt * 16 + n];
        }
        // online softmax per row (16 lanes of a half-wave share a row per VGPR)
        #pragma unroll
        for (int r = 0; r < 8; ++r) {
            float tm = red_max16(sacc[r]);
            float nm = fmaxf(run_max[r], tm);
            float corr = __expf(run_max[r] - nm);
            o[r] *= corr;
            run_sum[r] *= corr;
            float pe = __expf(sacc[r] - nm);
            run_sum[r] += red_sum16(pe);
            run_max[r] = nm;
            sP[h][(r + 8 * hi) * 16 + n] = (__bf16)pe;
        }
        __syncthreads();   // uniform trip count across all 8 waves

        // P fragment from LDS (K = j, 16 valid, padded to 32)
        v16bf ap;
        {
            v8bf lo = *(const v8bf*)(&sP[h][n * 16 + 8 * hi]);
            ap = make_v16(lo, z8);
        }
        // V fragment (transposed layout): one 16B vector load per lane
        v16bf bv = make_v16(*(const v8bf*)(v_t + ((size_t)(h * 16 + n)) * A_DIM + jt * 16 + 8 * hi), z8);
        o = wmma_bf16(ap, bv, o);
        __syncthreads();
    }

    #pragma unroll
    for (int r = 0; r < 8; ++r) {
        int m = it * 16 + r + 8 * hi;
        float val = (o[r] / (run_sum[r] + EPSF)) *
                    gate[(size_t)m * C_DIM + h * 16 + n];
        wa_bf[(size_t)m * C_DIM + h * 16 + n] = (__bf16)val;
    }
}

// ---------------------------------------------------------------------------
// Host launcher
// ---------------------------------------------------------------------------
extern "C" void kernel_launch(void* const* d_in, const int* in_sizes, int n_in,
                              void* d_out, int out_size, void* d_ws, size_t ws_size,
                              hipStream_t stream) {
    (void)in_sizes; (void)n_in; (void)out_size; (void)ws_size;
    const float* atom_act  = (const float*)d_in[0];
    const float* pair_act  = (const float*)d_in[1];
    const float* pair_mask = (const float*)d_in[3];
    const float* qn_s  = (const float*)d_in[4];
    const float* qn_b  = (const float*)d_in[5];
    const float* f2n_s = (const float*)d_in[6];
    const float* f2n_b = (const float*)d_in[7];
    const float* mn_s  = (const float*)d_in[8];
    const float* mn_b  = (const float*)d_in[9];
    const float* w2d   = (const float*)d_in[10];
    const float* w_pm  = (const float*)d_in[11];
    const float* b_pm  = (const float*)d_in[12];
    const float* wq    = (const float*)d_in[13];
    const float* wk    = (const float*)d_in[14];
    const float* wv    = (const float*)d_in[15];
    const float* wg    = (const float*)d_in[16];
    const float* bg    = (const float*)d_in[17];
    const float* wo    = (const float*)d_in[18];
    const float* bo    = (const float*)d_in[19];

    char* ws = (char*)d_ws;
    size_t off = 0;
    auto carve = [&](size_t bytes) -> void* {
        void* p = ws + off;
        off += (bytes + 255) & ~(size_t)255;
        return p;
    };
    const size_t AC = (size_t)A_DIM * C_DIM;

    float*  x        = (float*)carve(AC * 4);
    float*  u_acc    = (float*)carve(AC * 4);
    float*  v_acc    = (float*)carve(AC * 4);
    float*  msum0    = (float*)carve(A_DIM * 4);
    float*  msum1    = (float*)carve(A_DIM * 4);
    float*  gate     = (float*)carve(AC * 4);
    __bf16* qa_bf    = (__bf16*)carve(AC * 2);
    __bf16* ka_bf    = (__bf16*)carve(AC * 2);
    __bf16* q_bf     = (__bf16*)carve(AC * 2);
    __bf16* k_bf     = (__bf16*)carve(AC * 2);
    __bf16* v_t      = (__bf16*)carve(AC * 2);
    __bf16* wa_bf    = (__bf16*)carve(AC * 2);
    __bf16* w2d_f    = (__bf16*)carve(4 * 1 * 512 * 2);
    __bf16* wpm_f    = (__bf16*)carve(4 * 8 * 512 * 2);
    __bf16* wq_f     = (__bf16*)carve(4 * 8 * 512 * 2);
    __bf16* wk_f     = (__bf16*)carve(4 * 8 * 512 * 2);
    __bf16* wv_f     = (__bf16*)carve(4 * 8 * 512 * 2);
    __bf16* wg_f     = (__bf16*)carve(4 * 8 * 512 * 2);
    __bf16* wo_f     = (__bf16*)carve(4 * 8 * 512 * 2);
    float*  pair_bias = (float*)carve((size_t)H_DIM * A_DIM * A_DIM * 4);

    // zero accumulators (must be per-call: graph replays)
    zero_kernel<<<256, 256, 0, stream>>>(u_acc, (int)AC);
    zero_kernel<<<256, 256, 0, stream>>>(v_acc, (int)AC);
    zero_kernel<<<8,   256, 0, stream>>>(msum0, A_DIM);

    // weight swizzles
    swizzle_w_kernel<<<8,  256, 0, stream>>>(w2d, w2d_f, H_DIM, H_DIM, 1);
    swizzle_w_kernel<<<64, 256, 0, stream>>>(w_pm, wpm_f, C_DIM, C_DIM, 8);
    swizzle_w_kernel<<<64, 256, 0, stream>>>(wq,   wq_f,  C_DIM, C_DIM, 8);
    swizzle_w_kernel<<<64, 256, 0, stream>>>(wk,   wk_f,  C_DIM, C_DIM, 8);
    swizzle_w_kernel<<<64, 256, 0, stream>>>(wv,   wv_f,  C_DIM, C_DIM, 8);
    swizzle_w_kernel<<<64, 256, 0, stream>>>(wg,   wg_f,  C_DIM, C_DIM, 8);
    swizzle_w_kernel<<<64, 256, 0, stream>>>(wo,   wo_f,  C_DIM, C_DIM, 8);

    // query_norm(atom_act) and mask sums
    ln_x_kernel<<<64, 256, 0, stream>>>(atom_act, qn_s, qn_b, x);
    mask_sum_kernel<<<A_DIM, 256, 0, stream>>>(pair_mask, msum1, msum0);

    // the heavy fused pass over pair_act (4 tiles per block, async double-buffer)
    pair_kernel<<<dim3(A_DIM, A_DIM / 64), 256, 0, stream>>>(
        pair_act, pair_mask, f2n_s, f2n_b, mn_s, mn_b, b_pm,
        w2d_f, wpm_f, pair_bias, u_acc, v_acc);

    // query_act / key_act
    mix_kernel<<<(int)(AC / 256), 256, 0, stream>>>(x, u_acc, v_acc, msum1, msum0,
                                                    qa_bf, ka_bf);

    // projections: q (scaled by kph^-0.5 = 0.25), k, v (transposed), gate
    gemm_proj_kernel<<<64, 256, 0, stream>>>(qa_bf, wq_f, nullptr, 0.25f, 0, q_bf, nullptr, nullptr);
    gemm_proj_kernel<<<64, 256, 0, stream>>>(ka_bf, wk_f, nullptr, 1.0f,  0, k_bf, nullptr, nullptr);
    gemm_proj_kernel<<<64, 256, 0, stream>>>(ka_bf, wv_f, nullptr, 1.0f,  0, nullptr, v_t, nullptr);
    gemm_proj_kernel<<<64, 256, 0, stream>>>(qa_bf, wg_f, bg,      1.0f,  1, nullptr, nullptr, gate);

    // attention (flash-style), then output projection
    attn_kernel<<<64, 256, 0, stream>>>(q_bf, k_bf, v_t, pair_bias, gate, wa_bf);
    gemm_proj_kernel<<<64, 256, 0, stream>>>(wa_bf, wo_f, bo, 1.0f, 0, nullptr, nullptr,
                                             (float*)d_out);
}
